// NGCF_90881507983398
// MI455X (gfx1250) — matrix operand order repaired and checked by reading, not compiled, not run on hardware
//
#include <hip/hip_runtime.h>

typedef __attribute__((ext_vector_type(16))) _Float16 v16h;
typedef __attribute__((ext_vector_type(8)))  _Float16 v8h;
typedef __attribute__((ext_vector_type(8)))  float    v8f;

// ---------------------------------------------------------------------------
// f16-native WMMA GEMM: C[M,Nc] = act(A[M,K] @ W[K,Nc] (+bias) (+C))
// A is row-major f16 [M,K]; Wt is the TRANSPOSED weight, f16 [Nc,K], so both
// fragments load as contiguous vectors in the same k order (k = 16*half + s).
// FLAGS (compile-time): 1=bias, 2=accumulate f32 C, 4=relu,
//                       8=write f32 C, 16=write f16 Ch
// K % 64 == 0, M % 16 == 0, Nc % 16 == 0 (true for all model dims).
// ---------------------------------------------------------------------------
template<int FLAGS>
__global__ void __launch_bounds__(256)
gemm_wmma(const _Float16* __restrict__ A, const _Float16* __restrict__ Wt,
          const float* __restrict__ bias, float* __restrict__ C,
          _Float16* __restrict__ Ch, int M, int K, int Nc)
{
    const int lane   = threadIdx.x & 31;
    const int wave   = blockIdx.x * (blockDim.x >> 5) + (threadIdx.x >> 5);
    const int tilesN = Nc >> 4;
    const int totalT = (M >> 4) * tilesN;
    if (wave >= totalT) return;                 // uniform per wave: EXEC stays all-1s
    const int tileM = wave / tilesN;
    const int tileN = wave - tileM * tilesN;
    const int m0 = tileM << 4, n0 = tileN << 4;

    const int half = lane >> 4;                 // 0: lanes 0-15, 1: lanes 16-31
    const int l15  = lane & 15;
    const int kA   = half * 16;                 // A fragment K offset (contiguous k order, same as B)
    const int kB   = half * 16;                 // B fragment K offset (contiguous k order, same as A)

    const _Float16* arow = A  + (size_t)(m0 + l15) * K;
    const _Float16* brow = Wt + (size_t)(n0 + l15) * K;

    v8f acc0 = {0.f,0.f,0.f,0.f,0.f,0.f,0.f,0.f};
    v8f acc1 = {0.f,0.f,0.f,0.f,0.f,0.f,0.f,0.f};

    union AV { v16h v; v8h h[2]; };

    for (int k0 = 0; k0 < K; k0 += 64) {
        __builtin_prefetch(arow + k0 + 128, 0, 1);
        __builtin_prefetch(brow + k0 + 128, 0, 1);
        AV a0, a1;
        a0.h[0] = *(const v8h*)(arow + k0 + kA);        // K: kA .. kA+7
        a0.h[1] = *(const v8h*)(arow + k0 + 8 + kA);    // K: 8+kA .. 8+kA+7
        a1.h[0] = *(const v8h*)(arow + k0 + 32 + kA);
        a1.h[1] = *(const v8h*)(arow + k0 + 40 + kA);
        v16h b0 = *(const v16h*)(brow + k0 + kB);       // K: kB .. kB+15
        v16h b1 = *(const v16h*)(brow + k0 + 32 + kB);
        acc0 = __builtin_amdgcn_wmma_f32_16x16x32_f16(
                   false, a0.v, false, b0, (short)0, acc0, false, false);
        acc1 = __builtin_amdgcn_wmma_f32_16x16x32_f16(
                   false, a1.v, false, b1, (short)0, acc1, false, false);
    }

    const int cn = n0 + l15;
    const int rb = m0 + half * 8;               // C/D layout: VGPR r -> row rb+r
    float bv = (FLAGS & 1) ? bias[cn] : 0.0f;
#pragma unroll
    for (int r = 0; r < 8; ++r) {
        size_t idx = (size_t)(rb + r) * Nc + cn;
        float v = acc0[r] + acc1[r] + bv;
        if (FLAGS & 2)  v += C[idx];
        if (FLAGS & 4)  v = fmaxf(v, 0.0f);
        if (FLAGS & 8)  C[idx] = v;
        if (FLAGS & 16) Ch[idx] = (_Float16)v;
    }
}

// ---------------------------------------------------------------------------
// Weight transpose + f32->f16 convert: Wt[n*K + k] = (f16) W[k*Nc + n]
// ---------------------------------------------------------------------------
__global__ void wconv(const float* __restrict__ W, _Float16* __restrict__ Wt,
                      int K, int Nc)
{
    int t = blockIdx.x * blockDim.x + threadIdx.x;
    if (t >= K * Nc) return;
    int n = t / K, k = t - n * K;
    Wt[t] = (_Float16)W[(size_t)k * Nc + n];
}

__global__ void aconv(const float* __restrict__ X, _Float16* __restrict__ Xh, int n)
{
    int t = blockIdx.x * blockDim.x + threadIdx.x;
    if (t < n) Xh[t] = (_Float16)X[t];
}

// ---------------------------------------------------------------------------
// Graph / elementwise kernels
// ---------------------------------------------------------------------------
__global__ void gather_f16(const int* __restrict__ ui, const int* __restrict__ ii,
                           const float* __restrict__ ue, const float* __restrict__ ie,
                           _Float16* __restrict__ x0h, int N)
{
    int t = blockIdx.x * blockDim.x + threadIdx.x;
    if (t >= N * 128) return;
    int i = t >> 7, c = t & 127;
    float v = (c < 64) ? ue[(size_t)ui[i] * 64 + c]
                       : ie[(size_t)ii[i] * 64 + (c - 64)];
    x0h[t] = (_Float16)v;
}

__global__ void degree_kernel(const int* __restrict__ dst, float* __restrict__ cnt, int E)
{
    int t = blockIdx.x * blockDim.x + threadIdx.x;
    if (t < E) atomicAdd(cnt + dst[t], 1.0f);
}

__global__ void dinv_kernel(const float* __restrict__ cnt, float* __restrict__ dinv,
                            float* __restrict__ dinvc, int N)
{
    int i = blockIdx.x * blockDim.x + threadIdx.x;
    if (i >= N) return;
    float c = cnt[i];
    dinv[i]  = rsqrtf(c + 1.0f);                // GCN degree includes self-loop
    dinvc[i] = (c > 0.0f) ? rsqrtf(c) : 0.0f;   // Cheb degree (no self-loop)
}

// GCN aggregation with self-loops: edges [0,E) then N loop edges
__global__ void scatter_gcn(const int* __restrict__ src, const int* __restrict__ dst,
                            const float* __restrict__ dinv,
                            const float* __restrict__ Xin, float* __restrict__ Xout,
                            int E, int N, int C)
{
    int t = blockIdx.x * blockDim.x + threadIdx.x;
    if (t >= (E + N) * C) return;
    int e = t / C, c = t - e * C;
    int s, d;
    if (e < E) { s = src[e]; d = dst[e]; } else { s = d = e - E; }
    atomicAdd(Xout + (size_t)d * C + c, dinv[s] * dinv[d] * Xin[(size_t)s * C + c]);
}

// mode 0: plain sum (SAGE); mode 1: -dinvc[s]*dinvc[d] (Cheb Tx1); no self loops
__global__ void scatter_edges(const int* __restrict__ src, const int* __restrict__ dst,
                              const float* __restrict__ dinvc,
                              const float* __restrict__ Xin, float* __restrict__ Xout,
                              int E, int C, int mode)
{
    int t = blockIdx.x * blockDim.x + threadIdx.x;
    if (t >= E * C) return;
    int e = t / C, c = t - e * C;
    int s = src[e], d = dst[e];
    float coef = (mode == 1) ? (-dinvc[s] * dinvc[d]) : 1.0f;
    atomicAdd(Xout + (size_t)d * C + c, coef * Xin[(size_t)s * C + c]);
}

__global__ void addbias_kernel(float* __restrict__ X, const float* __restrict__ b,
                               int N, int C)
{
    int t = blockIdx.x * blockDim.x + threadIdx.x;
    if (t >= N * C) return;
    X[t] += b[t % C];
}

__global__ void colstats(const float* __restrict__ X, float* __restrict__ CS,
                         float* __restrict__ CQ, int N, int C)
{
    int c = blockIdx.x;
    float s = 0.f, q = 0.f;
    for (int i = threadIdx.x; i < N; i += blockDim.x) {
        float v = X[(size_t)i * C + c];
        s += v; q += v * v;
    }
    __shared__ float sh[256], shq[256];
    sh[threadIdx.x] = s; shq[threadIdx.x] = q;
    __syncthreads();
    for (int w = blockDim.x >> 1; w > 0; w >>= 1) {
        if (threadIdx.x < (unsigned)w) {
            sh[threadIdx.x]  += sh[threadIdx.x + w];
            shq[threadIdx.x] += shq[threadIdx.x + w];
        }
        __syncthreads();
    }
    if (threadIdx.x == 0) { CS[c] = sh[0]; CQ[c] = shq[0]; }
}

// OF: 1 = write f32, 2 = write f16 (compile-time specialized)
template<int OF>
__global__ void bn_relu(const float* __restrict__ Xin, float* __restrict__ Xout,
                        _Float16* __restrict__ Xh,
                        const float* __restrict__ CS, const float* __restrict__ CQ,
                        const float* __restrict__ g, const float* __restrict__ b,
                        int N, int C)
{
    int t = blockIdx.x * blockDim.x + threadIdx.x;
    if (t >= N * C) return;
    int c = t % C;
    float mu  = CS[c] / (float)N;
    float var = CQ[c] / (float)N - mu * mu;     // biased variance (jnp.var)
    float v = fmaxf((Xin[t] - mu) * rsqrtf(var + 1e-5f) * g[c] + b[c], 0.0f);
    if (OF & 1) Xout[t] = v;
    if (OF & 2) Xh[t]   = (_Float16)v;
}

__global__ void sage_div_f16(const float* __restrict__ agg, const float* __restrict__ cnt,
                             _Float16* __restrict__ outh, int N, int C)
{
    int t = blockIdx.x * blockDim.x + threadIdx.x;
    if (t >= N * C) return;
    outh[t] = (_Float16)(agg[t] / fmaxf(cnt[t / C], 1.0f));
}

__global__ void fillk(float* __restrict__ p, float v, int n)
{
    int i = blockIdx.x * blockDim.x + threadIdx.x;
    if (i < n) p[i] = v;
}

// GAT: per-node attention terms al/ar for H=2 heads, C=128
__global__ void gat_alar(const float* __restrict__ XP, const float* __restrict__ as_,
                         const float* __restrict__ ad_, float* __restrict__ AL,
                         float* __restrict__ AR, int N)
{
    int t = blockIdx.x * blockDim.x + threadIdx.x;
    if (t >= N * 2) return;
    int i = t >> 1, h = t & 1;
    const float* xp = XP + (size_t)i * 256 + h * 128;
    const float* vs = as_ + h * 128;
    const float* vd = ad_ + h * 128;
    float sl = 0.f, sr = 0.f;
    for (int c = 0; c < 128; ++c) { sl += xp[c] * vs[c]; sr += xp[c] * vd[c]; }
    AL[t] = sl; AR[t] = sr;
}

__device__ __forceinline__ void atomicMaxF(float* addr, float v)
{
    if (v >= 0.0f) atomicMax((int*)addr, __float_as_int(v));
    else           atomicMin((unsigned int*)addr, __float_as_uint(v));
}

__global__ void gat_edge1(const int* __restrict__ src, const int* __restrict__ dst,
                          const float* __restrict__ AL, const float* __restrict__ AR,
                          float* __restrict__ EE, float* __restrict__ MM, int E, int N)
{
    int t = blockIdx.x * blockDim.x + threadIdx.x;
    if (t >= (E + N) * 2) return;
    int e = t >> 1, h = t & 1;
    int s, d;
    if (e < E) { s = src[e]; d = dst[e]; } else { s = d = e - E; }
    float v = AL[s * 2 + h] + AR[d * 2 + h];
    v = (v > 0.0f) ? v : 0.2f * v;              // leaky_relu(0.2)
    EE[t] = v;
    atomicMaxF(MM + d * 2 + h, v);
}

__global__ void gat_edge2(const int* __restrict__ dst, float* __restrict__ EE,
                          const float* __restrict__ MM, float* __restrict__ DEN,
                          int E, int N)
{
    int t = blockIdx.x * blockDim.x + threadIdx.x;
    if (t >= (E + N) * 2) return;
    int e = t >> 1, h = t & 1;
    int d = (e < E) ? dst[e] : (e - E);
    float ex = __expf(EE[t] - MM[d * 2 + h]);
    EE[t] = ex;
    atomicAdd(DEN + d * 2 + h, ex);
}

__global__ void gat_edge3(const int* __restrict__ src, const int* __restrict__ dst,
                          const float* __restrict__ EE, const float* __restrict__ DEN,
                          const float* __restrict__ XP, float* __restrict__ OUT,
                          int E, int N)
{
    int t = blockIdx.x * blockDim.x + threadIdx.x;
    if (t >= (E + N) * 256) return;
    int e = t >> 8, rem = t & 255;
    int h = rem >> 7, c = rem & 127;
    int s, d;
    if (e < E) { s = src[e]; d = dst[e]; } else { s = d = e - E; }
    float alpha = EE[e * 2 + h] / DEN[d * 2 + h];
    atomicAdd(OUT + (size_t)d * 256 + h * 128 + c,
              alpha * XP[(size_t)s * 256 + h * 128 + c]);
}

// mean over heads + bias + elu -> f16 (feeds next GEMM / final projection)
__global__ void gat_combine(const float* __restrict__ OUT, const float* __restrict__ bg,
                            _Float16* __restrict__ Xh, int N)
{
    int t = blockIdx.x * blockDim.x + threadIdx.x;
    if (t >= N * 128) return;
    int i = t >> 7, c = t & 127;
    float v = 0.5f * (OUT[(size_t)i * 256 + c] + OUT[(size_t)i * 256 + 128 + c]) + bg[c];
    Xh[t] = (_Float16)((v > 0.0f) ? v : (__expf(v) - 1.0f));
}

__global__ void final_proj(const _Float16* __restrict__ Xh, const float* __restrict__ Wp,
                           const float* __restrict__ bp, float* __restrict__ out, int N)
{
    int i = blockIdx.x * blockDim.x + threadIdx.x;
    if (i >= N) return;
    float s = bp[0];
    const _Float16* x = Xh + (size_t)i * 128;
    for (int c = 0; c < 128; ++c) s += (float)x[c] * Wp[c];
    out[i] = s;
}

// ---------------------------------------------------------------------------
// Host-side orchestration
// ---------------------------------------------------------------------------
static inline dim3 grid1(long long total) { return dim3((unsigned)((total + 255) / 256)); }
static inline int gemm_blocks(int M, int Nc) { return ((M / 16) * (Nc / 16) + 7) / 8; }

extern "C" void kernel_launch(void* const* d_in, const int* in_sizes, int n_in,
                              void* d_out, int out_size, void* d_ws, size_t ws_size,
                              hipStream_t stream)
{
    const int N = in_sizes[0];
    const int E = in_sizes[2] / 2;

    const int*   user_idx = (const int*)d_in[0];
    const int*   item_idx = (const int*)d_in[1];
    const int*   src      = (const int*)d_in[2];
    const int*   dst      = src + E;
    const float* user_emb = (const float*)d_in[3];
    const float* item_emb = (const float*)d_in[4];
    const float* W1  = (const float*)d_in[5];  const float* b1  = (const float*)d_in[6];
    const float* W2  = (const float*)d_in[7];  const float* b2  = (const float*)d_in[8];
    const float* Wg1 = (const float*)d_in[9];  const float* bg1 = (const float*)d_in[10];
    const float* g1  = (const float*)d_in[11]; const float* be1 = (const float*)d_in[12];
    const float* Wg2 = (const float*)d_in[13]; const float* bg2 = (const float*)d_in[14];
    const float* g2  = (const float*)d_in[15]; const float* be2 = (const float*)d_in[16];
    const float* Wsl = (const float*)d_in[17]; const float* bsl = (const float*)d_in[18];
    const float* Wsr = (const float*)d_in[19];
    const float* Wc0 = (const float*)d_in[20]; const float* Wc1 = (const float*)d_in[21];
    const float* bc  = (const float*)d_in[22];
    const float* Wgat1 = (const float*)d_in[23]; const float* as1 = (const float*)d_in[24];
    const float* ad1   = (const float*)d_in[25]; const float* bgat1 = (const float*)d_in[26];
    const float* Wgat2 = (const float*)d_in[27]; const float* as2 = (const float*)d_in[28];
    const float* ad2   = (const float*)d_in[29]; const float* bgat2 = (const float*)d_in[30];
    const float* Wp  = (const float*)d_in[31]; const float* bp = (const float*)d_in[32];
    float* out = (float*)d_out;

    // ---- workspace carve (256B-aligned chunks)
    char* base = (char*)d_ws;
    size_t off = 0;
    auto carve = [&](size_t bytes) -> char* {
        char* q = base + off;
        off += (bytes + 255) & ~(size_t)255;
        return q;
    };
    // f32 buffers
    float* T1  = (float*)carve((size_t)N * 256 * 4);   // GCN1 xw / GAT xp
    float* T2  = (float*)carve((size_t)N * 256 * 4);   // GCN1 agg / GAT out
    float* T3  = (float*)carve((size_t)N * 128 * 4);   // SAGE agg / Cheb Tx1
    float* XA  = (float*)carve((size_t)N * 128 * 4);
    float* XB  = (float*)carve((size_t)N * 128 * 4);
    float* CNT   = (float*)carve((size_t)N * 4);
    float* DINV  = (float*)carve((size_t)N * 4);
    float* DINVC = (float*)carve((size_t)N * 4);
    float* CS = (float*)carve(1024 * 4);
    float* CQ = (float*)carve(1024 * 4);
    float* AL = (float*)carve((size_t)N * 2 * 4);
    float* AR = (float*)carve((size_t)N * 2 * 4);
    float* EE = (float*)carve((size_t)(E + N) * 2 * 4);
    float* MM = (float*)carve((size_t)N * 2 * 4);
    float* DEN = (float*)carve((size_t)N * 2 * 4);
    // f16 activation buffers
    _Float16* X0h = (_Float16*)carve((size_t)N * 128 * 2);
    _Float16* H1h = (_Float16*)carve((size_t)N * 1024 * 2);
    _Float16* H2h = (_Float16*)carve((size_t)N * 512 * 2);
    _Float16* T1h = (_Float16*)carve((size_t)N * 256 * 2);
    _Float16* G1  = (_Float16*)carve((size_t)N * 128 * 2);
    _Float16* G2  = (_Float16*)carve((size_t)N * 128 * 2);
    _Float16* G3  = (_Float16*)carve((size_t)N * 128 * 2);
    // f16 transposed weights [Nc,K]
    _Float16* W1t   = (_Float16*)carve((size_t)128 * 1024 * 2);
    _Float16* W2t   = (_Float16*)carve((size_t)1024 * 512 * 2);
    _Float16* Wg1t  = (_Float16*)carve((size_t)512 * 256 * 2);
    _Float16* Wg2t  = (_Float16*)carve((size_t)256 * 128 * 2);
    _Float16* Wslt  = (_Float16*)carve((size_t)128 * 128 * 2);
    _Float16* Wsrt  = (_Float16*)carve((size_t)128 * 128 * 2);
    _Float16* Wc0t  = (_Float16*)carve((size_t)128 * 128 * 2);
    _Float16* Wc1t  = (_Float16*)carve((size_t)128 * 128 * 2);
    _Float16* Wga1t = (_Float16*)carve((size_t)128 * 256 * 2);
    _Float16* Wga2t = (_Float16*)carve((size_t)128 * 256 * 2);

    // ---- weight transpose/convert (tiny; recomputed every call: deterministic)
    wconv<<<grid1(128 * 1024), 256, 0, stream>>>(W1, W1t, 128, 1024);
    wconv<<<grid1(1024 * 512), 256, 0, stream>>>(W2, W2t, 1024, 512);
    wconv<<<grid1(512 * 256), 256, 0, stream>>>(Wg1, Wg1t, 512, 256);
    wconv<<<grid1(256 * 128), 256, 0, stream>>>(Wg2, Wg2t, 256, 128);
    wconv<<<grid1(128 * 128), 256, 0, stream>>>(Wsl, Wslt, 128, 128);
    wconv<<<grid1(128 * 128), 256, 0, stream>>>(Wsr, Wsrt, 128, 128);
    wconv<<<grid1(128 * 128), 256, 0, stream>>>(Wc0, Wc0t, 128, 128);
    wconv<<<grid1(128 * 128), 256, 0, stream>>>(Wc1, Wc1t, 128, 128);
    wconv<<<grid1(128 * 256), 256, 0, stream>>>(Wgat1, Wga1t, 128, 256);
    wconv<<<grid1(128 * 256), 256, 0, stream>>>(Wgat2, Wga2t, 128, 256);

    // ---- degrees (shared by all graph layers)
    hipMemsetAsync(CNT, 0, (size_t)N * 4, stream);
    degree_kernel<<<grid1(E), 256, 0, stream>>>(dst, CNT, E);
    dinv_kernel<<<grid1(N), 256, 0, stream>>>(CNT, DINV, DINVC, N);

    // ---- embedding gather -> X0h f16 [N,128]
    gather_f16<<<grid1((long long)N * 128), 256, 0, stream>>>(
        user_idx, item_idx, user_emb, item_emb, X0h, N);

    // ---- MLP: H1h = relu(X0@W1+b1) f16; H2h = relu(H1@W2+b2) f16
    gemm_wmma<1|4|16><<<gemm_blocks(N, 1024), 256, 0, stream>>>(
        X0h, W1t, b1, nullptr, H1h, N, 128, 1024);
    gemm_wmma<1|4|16><<<gemm_blocks(N, 512), 256, 0, stream>>>(
        H1h, W2t, b2, nullptr, H2h, N, 1024, 512);

    // ---- GCN layer 1: 512 -> 256; sym-norm agg w/ self loops; +bias; BN; relu
    gemm_wmma<8><<<gemm_blocks(N, 256), 256, 0, stream>>>(
        H2h, Wg1t, nullptr, T1, nullptr, N, 512, 256);
    hipMemsetAsync(T2, 0, (size_t)N * 256 * 4, stream);
    scatter_gcn<<<grid1((long long)(E + N) * 256), 256, 0, stream>>>(
        src, dst, DINV, T1, T2, E, N, 256);
    addbias_kernel<<<grid1((long long)N * 256), 256, 0, stream>>>(T2, bg1, N, 256);
    colstats<<<256, 256, 0, stream>>>(T2, CS, CQ, N, 256);
    bn_relu<2><<<grid1((long long)N * 256), 256, 0, stream>>>(
        T2, nullptr, T1h, CS, CQ, g1, be1, N, 256);

    // ---- GCN layer 2: 256 -> 128
    gemm_wmma<8><<<gemm_blocks(N, 128), 256, 0, stream>>>(
        T1h, Wg2t, nullptr, XA, nullptr, N, 256, 128);
    hipMemsetAsync(XB, 0, (size_t)N * 128 * 4, stream);
    scatter_gcn<<<grid1((long long)(E + N) * 128), 256, 0, stream>>>(
        src, dst, DINV, XA, XB, E, N, 128);
    addbias_kernel<<<grid1((long long)N * 128), 256, 0, stream>>>(XB, bg2, N, 128);
    colstats<<<128, 256, 0, stream>>>(XB, CS, CQ, N, 128);
    bn_relu<3><<<grid1((long long)N * 128), 256, 0, stream>>>(
        XB, XA, G1, CS, CQ, g2, be2, N, 128);          // x: XA f32 + G1 f16

    // ---- SAGE: out = relu(mean_agg@Wsl + bsl + x@Wsr) -> XB f32 + G2 f16
    hipMemsetAsync(T3, 0, (size_t)N * 128 * 4, stream);
    scatter_edges<<<grid1((long long)E * 128), 256, 0, stream>>>(
        src, dst, DINVC, XA, T3, E, 128, 0);
    sage_div_f16<<<grid1((long long)N * 128), 256, 0, stream>>>(T3, CNT, G3, N, 128);
    gemm_wmma<1|8><<<gemm_blocks(N, 128), 256, 0, stream>>>(
        G3, Wslt, bsl, XB, nullptr, N, 128, 128);
    gemm_wmma<2|4|8|16><<<gemm_blocks(N, 128), 256, 0, stream>>>(
        G1, Wsrt, nullptr, XB, G2, N, 128, 128);

    // ---- Cheb K=2: out = relu(x@Wc0 + bc + Tx1@Wc1) -> G1 f16
    hipMemsetAsync(T3, 0, (size_t)N * 128 * 4, stream);
    scatter_edges<<<grid1((long long)E * 128), 256, 0, stream>>>(
        src, dst, DINVC, XB, T3, E, 128, 1);
    gemm_wmma<1|8><<<gemm_blocks(N, 128), 256, 0, stream>>>(
        G2, Wc0t, bc, XA, nullptr, N, 128, 128);
    aconv<<<grid1((long long)N * 128), 256, 0, stream>>>(T3, G3, N * 128);
    gemm_wmma<2|4|16><<<gemm_blocks(N, 128), 256, 0, stream>>>(
        G3, Wc1t, nullptr, XA, G1, N, 128, 128);

    // ---- GAT layers (H=2, C=128, concat=False -> mean over heads, +bias, elu)
    const _Float16* Wgt[2]  = {Wga1t, Wga2t};
    const float* as_[2]  = {as1, as2};
    const float* ad_[2]  = {ad1, ad2};
    const float* bgt[2]  = {bgat1, bgat2};
    _Float16* xin  = G1;
    _Float16* xout = G2;
    for (int l = 0; l < 2; ++l) {
        gemm_wmma<8><<<gemm_blocks(N, 256), 256, 0, stream>>>(
            xin, Wgt[l], nullptr, T1, nullptr, N, 128, 256);     // xp f32
        gat_alar<<<grid1((long long)N * 2), 256, 0, stream>>>(T1, as_[l], ad_[l], AL, AR, N);
        fillk<<<grid1((long long)N * 2), 256, 0, stream>>>(MM, -1e30f, N * 2);
        hipMemsetAsync(DEN, 0, (size_t)N * 2 * 4, stream);
        hipMemsetAsync(T2, 0, (size_t)N * 256 * 4, stream);
        gat_edge1<<<grid1((long long)(E + N) * 2), 256, 0, stream>>>(src, dst, AL, AR, EE, MM, E, N);
        gat_edge2<<<grid1((long long)(E + N) * 2), 256, 0, stream>>>(dst, EE, MM, DEN, E, N);
        gat_edge3<<<grid1((long long)(E + N) * 256), 256, 0, stream>>>(src, dst, EE, DEN, T1, T2, E, N);
        gat_combine<<<grid1((long long)N * 128), 256, 0, stream>>>(T2, bgt[l], xout, N);
        _Float16* tmp = xin; xin = xout; xout = tmp;
    }

    // ---- final projection: out = x@Wp + bp
    final_proj<<<grid1(N), 256, 0, stream>>>(xin, Wp, bp, out, N);
}
